// MultiScaleARSPBlock_6030134083931
// MI455X (gfx1250) — matrix operand order repaired and checked by reading, not compiled, not run on hardware
//
#include <hip/hip_runtime.h>

#define Nn    49152
#define Bg    64
#define MXN   768
#define Eg    491520
#define Dd    128
#define Cc    160
#define Tt    4
#define DIc   320
#define DSc   128
#define DTRc  10
#define CHR   64      // rows per attention-MLP chunk
#define NCH   12      // chunks per graph (768/64)

typedef __attribute__((ext_vector_type(16))) _Float16 v16h;
typedef __attribute__((ext_vector_type(8)))  float    v8f;

// ---------------------------------------------------------------- WMMA helpers
__device__ inline v8f wmma16(v16h a, v16h b, v8f c) {
  return __builtin_amdgcn_wmma_f32_16x16x32_f16(false, a, false, b, (short)0, c, false, false);
}

// A fragment (16x32 f16), row-major source, ptr at tile origin. ISA 7.12.2 layout:
// lanes 0-15: M=lane, K pairs {0,1..6,7} in v0-3, {16..23} in v4-7; lanes 16-31: K+8 / K+24.
__device__ inline v16h frag_a_f16(const _Float16* A, int lda, int lane) {
  int m = lane & 15, kh = (lane >> 4) << 3;
  v16h r;
#pragma unroll
  for (int i = 0; i < 8; ++i) {
    int k = ((i >> 2) << 4) + kh + ((i & 3) << 1);
    r[2 * i]     = A[m * lda + k];
    r[2 * i + 1] = A[m * lda + k + 1];
  }
  return r;
}

// B fragment (32x16 f16) from TRANSPOSED (N-major) weights: Bt[n*ldk + k].
// lane n = lane&15, K = (lane>>4)*8 + {0..7} then 16 + (lane>>4)*8 + {0..7} (matches A layout).
__device__ inline v16h frag_b_f16t(const _Float16* Bt, int ldk, int lane) {
  const _Float16* p = Bt + (lane & 15) * ldk + ((lane >> 4) << 3);
  v16h r;
#pragma unroll
  for (int i = 0; i < 16; ++i) r[i] = p[((i >> 3) << 4) + (i & 7)];
  return r;
}

__device__ inline float sigm(float v) { return 1.f / (1.f + expf(-v)); }
__device__ inline float silu_(float v) { return v * sigm(v); }

// ---------------------------------------------------------------- small kernels
// f32 (K x N, row-major) -> f16 transposed (N x K)
__global__ void cvt16t_kernel(const float* __restrict__ s, _Float16* __restrict__ d,
                              int K, int N) {
  int i = blockIdx.x * blockDim.x + threadIdx.x;
  if (i >= K * N) return;
  int k = i / N, n = i % N;
  d[n * K + k] = (_Float16)s[i];
}

__global__ void prep_xd_kernel(const float* __restrict__ x, const float* __restrict__ lpe,
                               const float* __restrict__ rwse, _Float16* __restrict__ xd) {
  long long t = (long long)blockIdx.x * blockDim.x + threadIdx.x;
  int n = (int)(t / Cc), c = (int)(t % Cc);
  if (n >= Nn) return;
  float v = (c < Dd) ? x[(size_t)n * Dd + c]
                     : ((c < Dd + 16) ? lpe[(size_t)n * 16 + (c - Dd)]
                                      : rwse[(size_t)n * 16 + (c - Dd - 16)]);
  xd[(size_t)n * Cc + c] = (_Float16)v;
}

// gate = sigmoid(edge_attr @ We + be) ; agg[dst] += x[src] * gate
// one lane per (edge, 4 features): b128 gather + 4 atomics
__global__ void edge_kernel(const float* __restrict__ x, const float* __restrict__ ea,
                            const int* __restrict__ ei, const float* __restrict__ We,
                            const float* __restrict__ be, float* __restrict__ agg) {
  long long t = (long long)blockIdx.x * blockDim.x + threadIdx.x;
  int e = (int)(t >> 5);
  if (e >= Eg) return;
  int j = (int)(t & 31) << 2;
  float a0 = ea[(size_t)e * 2], a1 = ea[(size_t)e * 2 + 1];
  int src = ei[e], dst = ei[Eg + e];
  const float4 w0 = *(const float4*)(We + j);
  const float4 w1 = *(const float4*)(We + Dd + j);
  const float4 bb = *(const float4*)(be + j);
  const float4 xv = *(const float4*)(x + (size_t)src * Dd + j);
  float* ap = agg + (size_t)dst * Dd + j;
  atomicAdd(ap + 0, xv.x * sigm(a0 * w0.x + a1 * w1.x + bb.x));
  atomicAdd(ap + 1, xv.y * sigm(a0 * w0.y + a1 * w1.y + bb.y));
  atomicAdd(ap + 2, xv.z * sigm(a0 * w0.z + a1 * w1.z + bb.z));
  atomicAdd(ap + 3, xv.w * sigm(a0 * w0.w + a1 * w1.w + bb.w));
}

// xs_i = relu((x + agg) @ Wx_i + bx_i) -> f16 columns of cat (N x 544)
// A panel in LDS; C tiles bounced through LDS for coalesced b32 row stores
__global__ __launch_bounds__(256) void gemm_xs_kernel(
    const float* __restrict__ x, const float* __restrict__ agg,
    const _Float16* __restrict__ Wt, const float* __restrict__ bias,
    _Float16* __restrict__ cat, int colOff) {
  __shared__ _Float16 ldsA[16 * Dd];
  __shared__ _Float16 ldsO[16 * Dd];
  int tid = threadIdx.x, wave = tid >> 5, lane = tid & 31;
  int m0 = blockIdx.x * 16;
  for (int i = tid; i < 16 * Dd; i += 256) {
    size_t g = (size_t)(m0 + (i >> 7)) * Dd + (i & 127);
    ldsA[i] = (_Float16)(x[g] + agg[g]);
  }
  __syncthreads();
  int n0 = wave << 4;
  float bv = bias[n0 + (lane & 15)];
  v8f c;
#pragma unroll
  for (int v = 0; v < 8; ++v) c[v] = bv;
#pragma unroll
  for (int k0 = 0; k0 < Dd; k0 += 32)
    c = wmma16(frag_a_f16(ldsA + k0, Dd, lane),
               frag_b_f16t(Wt + n0 * Dd + k0, Dd, lane), c);
  int col = n0 + (lane & 15);
  int rb = (lane >> 4) << 3;
#pragma unroll
  for (int v = 0; v < 8; ++v)
    ldsO[(rb + v) * Dd + col] = (_Float16)fmaxf(c[v], 0.f);
  __syncthreads();
  const unsigned int* src = (const unsigned int*)ldsO;
  for (int i = tid; i < 16 * 64; i += 256) {   // 16 rows x 64 dwords, coalesced
    int r = i >> 6, dw = i & 63;
    *(unsigned int*)(cat + (size_t)(m0 + r) * 544 + colOff + (dw << 1)) = src[i];
  }
}

// one 160->160 relu MLP layer over a 64-row chunk, WMMA tiled (4x10 tiles / 8 waves)
__device__ inline void mlp64(const _Float16* A, const _Float16* Wt,
                             const float* bias, _Float16* out,
                             int wave, int lane) {
  for (int tile = wave; tile < 40; tile += 8) {
    int mt = tile / 10, nt = tile % 10;
    int m0 = mt << 4, n0 = nt << 4;
    float bv = bias[n0 + (lane & 15)];
    v8f c;
#pragma unroll
    for (int v = 0; v < 8; ++v) c[v] = bv;
#pragma unroll
    for (int k0 = 0; k0 < Cc; k0 += 32)
      c = wmma16(frag_a_f16(A + m0 * Cc + k0, Cc, lane),
                 frag_b_f16t(Wt + n0 * Cc + k0, Cc, lane), c);
    int col = n0 + (lane & 15);
    int rb = m0 + ((lane >> 4) << 3);
#pragma unroll
    for (int v = 0; v < 8; ++v)
      out[(rb + v) * Cc + col] = (_Float16)fmaxf(c[v], 0.f);
  }
}

// hbias[b] = wm_b0 + h[b] @ wm_w0[160:,:]   (64 blocks x 160 threads)
__global__ void hbias_kernel(const float* __restrict__ h, const float* __restrict__ w0,
                             const float* __restrict__ b0, float* __restrict__ hb) {
  __shared__ float hl[Cc];
  int b = blockIdx.x, j = threadIdx.x;
  hl[j] = h[b * Cc + j];
  __syncthreads();
  float acc = b0[j];
  for (int k = 0; k < Cc; ++k) acc += hl[k] * w0[(Cc + k) * Cc + j];
  hb[b * Cc + j] = acc;
}

// attention MLP over one 64-row chunk of one graph (64*12 = 768 blocks)
__global__ __launch_bounds__(256) void attn_mlp_kernel(
    const _Float16* __restrict__ xd, const _Float16* __restrict__ w0a,
    const _Float16* __restrict__ w1, const _Float16* __restrict__ w2,
    const float* __restrict__ hbias, const float* __restrict__ wm_b1,
    const float* __restrict__ wm_b2, const float* __restrict__ wm_w3,
    const float* __restrict__ wm_b3, float* __restrict__ scores_g) {
  extern __shared__ char smem_raw[];
  _Float16* bufX = (_Float16*)smem_raw;        // 64*160 f16
  _Float16* bufA = bufX + CHR * Cc;            // 64*160 f16
  _Float16* bufB = bufA + CHR * Cc;            // 64*160 f16
  int blk = blockIdx.x;
  int b = blk / NCH, ch = blk % NCH;
  int tid = threadIdx.x, wave = tid >> 5, lane = tid & 31;
  const _Float16* xg = xd + ((size_t)b * MXN + ch * CHR) * Cc;
  // stage chunk into LDS (b32 copies)
  const unsigned int* xsrc = (const unsigned int*)xg;
  for (int i = tid; i < (CHR * Cc) / 2; i += 256) ((unsigned int*)bufX)[i] = xsrc[i];
  // warm next layer's weights into cache while layer0 runs
  __builtin_prefetch((const char*)w1 + tid * 200, 0, 1);
  __syncthreads();
  mlp64(bufX, w0a, hbias + b * Cc, bufA, wave, lane);
  __syncthreads();
  mlp64(bufA, w1, wm_b1, bufB, wave, lane);
  __syncthreads();
  mlp64(bufB, w2, wm_b2, bufA, wave, lane);
  __syncthreads();
  if (tid < CHR) {
    float acc = wm_b3[0];
    for (int k = 0; k < Cc; ++k) acc += (float)bufA[tid * Cc + k] * wm_w3[k];
    scores_g[b * MXN + ch * CHR + tid] = acc * 0.08838834764831845f;  // 128^-0.5
  }
}

__device__ inline float blk_red(float v, float* red, int tid, bool do_max) {
#pragma unroll
  for (int o = 16; o > 0; o >>= 1) {
    float w = __shfl_down(v, o, 32);
    v = do_max ? fmaxf(v, w) : (v + w);
  }
  if ((tid & 31) == 0) red[tid >> 5] = v;
  __syncthreads();
  if (tid == 0) {
    float r = red[0];
    for (int i = 1; i < 8; ++i) r = do_max ? fmaxf(r, red[i]) : (r + red[i]);
    red[0] = r;
  }
  __syncthreads();
  float r = red[0];
  __syncthreads();
  return r;
}

// softmax + weighted mean + GRU per graph (64 blocks)
__global__ __launch_bounds__(256) void attn_reduce_kernel(
    const _Float16* __restrict__ xd, const float* __restrict__ scores_g,
    const float* __restrict__ gWi, const float* __restrict__ gWh,
    const float* __restrict__ gbi, const float* __restrict__ gbh,
    float* __restrict__ h, float* __restrict__ tokens, int step) {
  __shared__ float scores[MXN];
  __shared__ float hl[Cc], tv[Cc];
  __shared__ float gg[960];
  __shared__ float red[16];
  int b = blockIdx.x, tid = threadIdx.x;
  for (int i = tid; i < MXN; i += 256) scores[i] = scores_g[b * MXN + i];
  if (tid < Cc) hl[tid] = h[b * Cc + tid];
  __syncthreads();
  float mloc = -3.4e38f;
  for (int i = tid; i < MXN; i += 256) mloc = fmaxf(mloc, scores[i]);
  float mx = blk_red(mloc, red, tid, true);
  float sloc = 0.f;
  for (int i = tid; i < MXN; i += 256) {
    float e = expf(scores[i] - mx);
    scores[i] = e;
    sloc += e;
  }
  float ssum = blk_red(sloc, red, tid, false);
  float inv = 1.f / ssum;
  __syncthreads();
  const _Float16* xg = xd + (size_t)b * MXN * Cc;
  for (int c0 = tid; c0 < Cc; c0 += 256) {
    float acc = 0.f;
    for (int r = 0; r < MXN; ++r) acc += scores[r] * (float)xg[(size_t)r * Cc + c0];
    float tval = acc * inv * (1.f / (float)MXN);
    tv[c0] = tval;
    tokens[((size_t)b * Tt + step) * Cc + c0] = tval;
  }
  __syncthreads();
  for (int j = tid; j < 480; j += 256) {
    float gi = gbi[j], gh = gbh[j];
    for (int k = 0; k < Cc; ++k) {
      gi += tv[k] * gWi[k * 480 + j];
      gh += hl[k] * gWh[k * 480 + j];
    }
    gg[j] = gi;
    scores[j] = gh;   // reuse scores as gh store
  }
  __syncthreads();
  if (tid < Cc) {
    float ir = gg[tid], iz = gg[Cc + tid], inn = gg[2 * Cc + tid];
    float hr = scores[tid], hz = scores[Cc + tid], hn = scores[2 * Cc + tid];
    float r = sigm(ir + hr);
    float z = sigm(iz + hz);
    float n = tanhf(inn + r * hn);
    h[b * Cc + tid] = (1.f - z) * n + z * hl[tid];
  }
}

// Mamba-style SSM over T=4 tokens; one block per graph, 320 threads
__global__ __launch_bounds__(320) void mamba_kernel(
    const float* __restrict__ tokens, const float* __restrict__ Win,
    const float* __restrict__ convw, const float* __restrict__ convb,
    const float* __restrict__ Wx, const float* __restrict__ Wdt,
    const float* __restrict__ bdt, const float* __restrict__ Alog,
    const float* __restrict__ Dp, const float* __restrict__ Wout,
    const float* __restrict__ lg, const float* __restrict__ lb,
    float* __restrict__ mo_last) {
  __shared__ float tok[Tt * Cc];
  __shared__ float xz[Tt * 2 * DIc];
  __shared__ float xmc[Tt * DIc];
  __shared__ float dtrb[Tt * DTRc];
  __shared__ float Bm[Tt * DSc];
  __shared__ float Cm[Tt * DSc];
  __shared__ float dtv[Tt * DIc];
  __shared__ float ylast[DIc];
  __shared__ float mo3[Cc];
  __shared__ float stats[2];
  int b = blockIdx.x, tid = threadIdx.x;

  for (int i = tid; i < Tt * Cc; i += 320) tok[i] = tokens[(size_t)b * Tt * Cc + i];
  __syncthreads();
  for (int i = tid; i < Tt * 2 * DIc; i += 320) {
    int t = i / (2 * DIc), col = i % (2 * DIc);
    float acc = 0.f;
    for (int k = 0; k < Cc; ++k) acc += tok[t * Cc + k] * Win[k * (2 * DIc) + col];
    xz[i] = acc;
  }
  __syncthreads();
  for (int i = tid; i < Tt * DIc; i += 320) {
    int t = i / DIc, d = i % DIc;
    float acc = convb[d];
#pragma unroll
    for (int k = 0; k < 4; ++k) {
      int tq = t + k - 3;
      if (tq >= 0) acc += xz[tq * 2 * DIc + d] * convw[d * 4 + k];
    }
    xmc[i] = silu_(acc);
  }
  __syncthreads();
  for (int i = tid; i < Tt * 266; i += 320) {
    int t = i / 266, j = i % 266;
    float acc = 0.f;
    for (int d = 0; d < DIc; ++d) acc += xmc[t * DIc + d] * Wx[d * 266 + j];
    if (j < DTRc) dtrb[t * DTRc + j] = acc;
    else if (j < DTRc + DSc) Bm[t * DSc + (j - DTRc)] = acc;
    else Cm[t * DSc + (j - DTRc - DSc)] = acc;
  }
  __syncthreads();
  for (int i = tid; i < Tt * DIc; i += 320) {
    int t = i / DIc, d = i % DIc;
    float acc = bdt[d];
    for (int j = 0; j < DTRc; ++j) acc += dtrb[t * DTRc + j] * Wdt[j * DIc + d];
    dtv[i] = (acc > 25.f) ? acc : logf(1.f + expf(acc));
  }
  __syncthreads();
  if (tid < DIc) {
    int d = tid;
    float s_state[DSc];
#pragma unroll
    for (int ss = 0; ss < DSc; ++ss) s_state[ss] = 0.f;
    float yfin = 0.f;
    for (int t = 0; t < Tt; ++t) {
      float dtd = dtv[t * DIc + d];
      float xd = xmc[t * DIc + d];
      float acc = 0.f;
#pragma unroll 16
      for (int ss = 0; ss < DSc; ++ss) {
        float Av = expf(Alog[(size_t)d * DSc + ss]);          // A = -exp(Alog)
        s_state[ss] = s_state[ss] * expf(-dtd * Av) + dtd * Bm[t * DSc + ss] * xd;
        acc += s_state[ss] * Cm[t * DSc + ss];
      }
      if (t == Tt - 1) {
        float y = acc + Dp[d] * xd;
        yfin = y * silu_(xz[t * 2 * DIc + DIc + d]);          // gate by silu(z)
      }
    }
    ylast[d] = yfin;
  }
  __syncthreads();
  if (tid < Cc) {
    float acc = 0.f;
    for (int d = 0; d < DIc; ++d) acc += ylast[d] * Wout[d * Cc + tid];
    mo3[tid] = acc;
  }
  __syncthreads();
  if (tid == 0) {
    float mu = 0.f;
    for (int j = 0; j < Cc; ++j) mu += mo3[j];
    mu /= Cc;
    float var = 0.f;
    for (int j = 0; j < Cc; ++j) { float dd = mo3[j] - mu; var += dd * dd; }
    var /= Cc;
    stats[0] = mu;
    stats[1] = rsqrtf(var + 1e-5f);
  }
  __syncthreads();
  if (tid < Cc)
    mo_last[b * Cc + tid] = (mo3[tid] - stats[0]) * stats[1] * lg[tid] + lb[tid];
}

__global__ void scatter_mo_kernel(const float* __restrict__ mo_last,
                                  const int* __restrict__ batch,
                                  _Float16* __restrict__ cat) {
  long long t = (long long)blockIdx.x * blockDim.x + threadIdx.x;
  int n = (int)(t / Cc), c = (int)(t % Cc);
  if (n >= Nn) return;
  cat[(size_t)n * 544 + 384 + c] = (_Float16)mo_last[(size_t)batch[n] * Cc + c];
}

// fused: hmid = relu(LN(cat@mg_w0+b0)); x_new = hmid@mg_w1+b1; out = LN(x + x_new)
__global__ __launch_bounds__(256) void merge_kernel(
    const _Float16* __restrict__ cat, const _Float16* __restrict__ W0t,
    const _Float16* __restrict__ W1t, const float* __restrict__ b0,
    const float* __restrict__ g0, const float* __restrict__ beta0,
    const float* __restrict__ b1, const float* __restrict__ x,
    const float* __restrict__ lng, const float* __restrict__ lnb,
    float* __restrict__ out) {
  __shared__ _Float16 aLds[16 * 544];
  __shared__ float c1[16 * 128];
  __shared__ _Float16 h16[16 * 128];
  __shared__ float c2[16 * 128];
  __shared__ float mu1[16], rs1[16];
  int tid = threadIdx.x, wave = tid >> 5, lane = tid & 31;
  int m0 = blockIdx.x * 16;
  int n0 = wave * 16;
  {
    const unsigned int* src = (const unsigned int*)(cat + (size_t)m0 * 544);
    for (int i = tid; i < (16 * 544) / 2; i += 256) ((unsigned int*)aLds)[i] = src[i];
  }
  __syncthreads();
  {
    float bv = b0[n0 + (lane & 15)];
    v8f c;
#pragma unroll
    for (int v = 0; v < 8; ++v) c[v] = bv;
    for (int k0 = 0; k0 < 544; k0 += 32)
      c = wmma16(frag_a_f16(aLds + k0, 544, lane),
                 frag_b_f16t(W0t + n0 * 544 + k0, 544, lane), c);
    int col = n0 + (lane & 15), rb = (lane >> 4) << 3;
#pragma unroll
    for (int v = 0; v < 8; ++v) c1[(rb + v) * 128 + col] = c[v];
  }
  __syncthreads();
  if (tid < 16) {
    float mu = 0.f;
    for (int j = 0; j < 128; ++j) mu += c1[tid * 128 + j];
    mu *= (1.f / 128.f);
    float var = 0.f;
    for (int j = 0; j < 128; ++j) { float dd = c1[tid * 128 + j] - mu; var += dd * dd; }
    mu1[tid] = mu;
    rs1[tid] = rsqrtf(var * (1.f / 128.f) + 1e-5f);
  }
  __syncthreads();
  for (int i = tid; i < 2048; i += 256) {
    int r = i >> 7, cc = i & 127;
    float v = (c1[i] - mu1[r]) * rs1[r] * g0[cc] + beta0[cc];
    h16[i] = (_Float16)fmaxf(v, 0.f);
  }
  __syncthreads();
  {
    float bv = b1[n0 + (lane & 15)];
    v8f c;
#pragma unroll
    for (int v = 0; v < 8; ++v) c[v] = bv;
#pragma unroll
    for (int k0 = 0; k0 < 128; k0 += 32)
      c = wmma16(frag_a_f16(h16 + k0, 128, lane),
                 frag_b_f16t(W1t + n0 * 128 + k0, 128, lane), c);
    int col = n0 + (lane & 15), rb = (lane >> 4) << 3;
#pragma unroll
    for (int v = 0; v < 8; ++v) c2[(rb + v) * 128 + col] = c[v];
  }
  __syncthreads();
  for (int i = tid; i < 2048; i += 256)
    c2[i] += x[(size_t)(m0 + (i >> 7)) * 128 + (i & 127)];
  __syncthreads();
  if (tid < 16) {
    float mu = 0.f;
    for (int j = 0; j < 128; ++j) mu += c2[tid * 128 + j];
    mu *= (1.f / 128.f);
    float var = 0.f;
    for (int j = 0; j < 128; ++j) { float dd = c2[tid * 128 + j] - mu; var += dd * dd; }
    mu1[tid] = mu;
    rs1[tid] = rsqrtf(var * (1.f / 128.f) + 1e-5f);
  }
  __syncthreads();
  for (int i = tid; i < 2048; i += 256) {
    int r = i >> 7, cc = i & 127;
    out[(size_t)(m0 + r) * 128 + cc] = (c2[i] - mu1[r]) * rs1[r] * lng[cc] + lnb[cc];
  }
}

// ---------------------------------------------------------------- host
extern "C" void kernel_launch(void* const* d_in, const int* in_sizes, int n_in,
                              void* d_out, int out_size, void* d_ws, size_t ws_size,
                              hipStream_t stream) {
  (void)in_sizes; (void)n_in; (void)out_size; (void)ws_size;
  const float* x       = (const float*)d_in[0];
  const float* ea      = (const float*)d_in[1];
  const float* lpe     = (const float*)d_in[2];
  const float* rwse    = (const float*)d_in[3];
  const int*   ei      = (const int*)d_in[4];
  const int*   batch   = (const int*)d_in[5];
  const float* sc_We   = (const float*)d_in[6];
  const float* sc_be   = (const float*)d_in[7];
  const float* sc_Wx   = (const float*)d_in[8];
  const float* sc_bx   = (const float*)d_in[9];
  const float* wm_w0   = (const float*)d_in[10];
  const float* wm_b0   = (const float*)d_in[11];
  const float* wm_w1   = (const float*)d_in[12];
  const float* wm_b1   = (const float*)d_in[13];
  const float* wm_w2   = (const float*)d_in[14];
  const float* wm_b2   = (const float*)d_in[15];
  const float* wm_w3   = (const float*)d_in[16];
  const float* wm_b3   = (const float*)d_in[17];
  const float* gru_Wi  = (const float*)d_in[18];
  const float* gru_Wh  = (const float*)d_in[19];
  const float* gru_bi  = (const float*)d_in[20];
  const float* gru_bh  = (const float*)d_in[21];
  const float* m_Win   = (const float*)d_in[22];
  const float* m_convw = (const float*)d_in[23];
  const float* m_convb = (const float*)d_in[24];
  const float* m_Wx    = (const float*)d_in[25];
  const float* m_Wdt   = (const float*)d_in[26];
  const float* m_bdt   = (const float*)d_in[27];
  const float* m_Alog  = (const float*)d_in[28];
  const float* m_Dp    = (const float*)d_in[29];
  const float* m_Wout  = (const float*)d_in[30];
  const float* lnm_g   = (const float*)d_in[31];
  const float* lnm_b   = (const float*)d_in[32];
  const float* mg_w0   = (const float*)d_in[33];
  const float* mg_b0   = (const float*)d_in[34];
  const float* mg_g0   = (const float*)d_in[35];
  const float* mg_beta0= (const float*)d_in[36];
  const float* mg_w1   = (const float*)d_in[37];
  const float* mg_b1   = (const float*)d_in[38];
  const float* ln_g    = (const float*)d_in[39];
  const float* ln_b    = (const float*)d_in[40];
  float* out = (float*)d_out;

  char* ws = (char*)d_ws;
  size_t off = 0;
  auto alloc = [&](size_t bytes) {
    void* p = ws + off;
    off = (off + bytes + 255) & ~(size_t)255;
    return p;
  };
  float*    agg     = (float*)alloc((size_t)Nn * Dd * 4);
  _Float16* cat16   = (_Float16*)alloc((size_t)Nn * 544 * 2);
  _Float16* xd16    = (_Float16*)alloc((size_t)Nn * Cc * 2);
  _Float16* w0a16   = (_Float16*)alloc(25600 * 2);   // transposed 160x160
  _Float16* w1_16   = (_Float16*)alloc(25600 * 2);
  _Float16* w2_16   = (_Float16*)alloc(25600 * 2);
  _Float16* scWx16  = (_Float16*)alloc(49152 * 2);   // 3x transposed 128x128
  _Float16* mgw0_16 = (_Float16*)alloc(69632 * 2);   // transposed 128x544
  _Float16* mgw1_16 = (_Float16*)alloc(16384 * 2);
  float*    hbuf    = (float*)alloc(Bg * Cc * 4);
  float*    hbias   = (float*)alloc(Bg * Cc * 4);
  float*    scoresW = (float*)alloc((size_t)Bg * MXN * 4);
  float*    tokens  = (float*)alloc((size_t)Bg * Tt * Cc * 4);
  float*    mo_last = (float*)alloc(Bg * Cc * 4);

  // weight conversions (f32 row-major -> f16 N-major/transposed)
  cvt16t_kernel<<<(25600 + 255) / 256, 256, 0, stream>>>(wm_w0, w0a16, 160, 160);
  cvt16t_kernel<<<(25600 + 255) / 256, 256, 0, stream>>>(wm_w1, w1_16, 160, 160);
  cvt16t_kernel<<<(25600 + 255) / 256, 256, 0, stream>>>(wm_w2, w2_16, 160, 160);
  for (int i = 0; i < 3; ++i)
    cvt16t_kernel<<<(16384 + 255) / 256, 256, 0, stream>>>(
        sc_Wx + i * 16384, scWx16 + i * 16384, 128, 128);
  cvt16t_kernel<<<(69632 + 255) / 256, 256, 0, stream>>>(mg_w0, mgw0_16, 544, 128);
  cvt16t_kernel<<<(16384 + 255) / 256, 256, 0, stream>>>(mg_w1, mgw1_16, 128, 128);

  prep_xd_kernel<<<((size_t)Nn * Cc + 255) / 256, 256, 0, stream>>>(x, lpe, rwse, xd16);
  hipMemsetAsync(hbuf, 0, (size_t)Bg * Cc * 4, stream);

  // three gated-conv scales (sequential, reuse agg buffer)
  for (int i = 0; i < 3; ++i) {
    hipMemsetAsync(agg, 0, (size_t)Nn * Dd * 4, stream);
    edge_kernel<<<((size_t)Eg * 32) / 256, 256, 0, stream>>>(
        x, ea, ei, sc_We + i * 2 * Dd, sc_be + i * Dd, agg);
    gemm_xs_kernel<<<Nn / 16, 256, 0, stream>>>(
        x, agg, scWx16 + i * Dd * Dd, sc_bx + i * Dd, cat16, i * Dd);
  }

  // T=4 attention + GRU iterations (sequential dependence via h)
  size_t mlp_smem = (size_t)(3 * CHR * Cc * 2);   // bufX+bufA+bufB = 60 KB
  for (int s = 0; s < Tt; ++s) {
    hbias_kernel<<<Bg, Cc, 0, stream>>>(hbuf, wm_w0, wm_b0, hbias);
    attn_mlp_kernel<<<Bg * NCH, 256, mlp_smem, stream>>>(
        xd16, w0a16, w1_16, w2_16, hbias, wm_b1, wm_b2, wm_w3, wm_b3, scoresW);
    attn_reduce_kernel<<<Bg, 256, 0, stream>>>(
        xd16, scoresW, gru_Wi, gru_Wh, gru_bi, gru_bh, hbuf, tokens, s);
  }

  mamba_kernel<<<Bg, 320, 0, stream>>>(tokens, m_Win, m_convw, m_convb, m_Wx,
                                       m_Wdt, m_bdt, m_Alog, m_Dp, m_Wout,
                                       lnm_g, lnm_b, mo_last);

  scatter_mo_kernel<<<((size_t)Nn * Cc + 255) / 256, 256, 0, stream>>>(mo_last, batch, cat16);

  merge_kernel<<<Nn / 16, 256, 0, stream>>>(cat16, mgw0_16, mgw1_16, mg_b0, mg_g0,
                                            mg_beta0, mg_b1, x, ln_g, ln_b, out);
}
